// MultiHeadAttention_68204080661228
// MI455X (gfx1250) — matrix operand order repaired and checked by reading, not compiled, not run on hardware
//
#include <hip/hip_runtime.h>

// ---------- types for CDNA5 WMMA (gfx1250, wave32) ----------
typedef __attribute__((ext_vector_type(16))) __bf16        v16bf;
typedef __attribute__((ext_vector_type(8)))  float         v8f;
typedef __attribute__((ext_vector_type(4)))  unsigned int  u32x4;

union FragU {
    v16bf v;
    u32x4 q[2];
};

#define DIMC     1024
#define NSEQ     2048
#define NROWS    4096          // B*N
#define HEADS    16
#define HEAD_DIM 64
#define SCALE_C  0.125f        // 64^-0.5
#define LOG2E    1.44269504088896f

__device__ __forceinline__ unsigned short bf16_rne(float f) {
    unsigned int u = __float_as_uint(f);
    u += 0x7fffu + ((u >> 16) & 1u);
    return (unsigned short)(u >> 16);
}

__device__ __forceinline__ v8f zero8() {
    v8f z;
#pragma unroll
    for (int i = 0; i < 8; ++i) z[i] = 0.0f;
    return z;
}

__device__ __forceinline__ v8f wmma_bf16(const v16bf& a, const v16bf& b, const v8f& c) {
    return __builtin_amdgcn_wmma_f32_16x16x32_bf16(
        false, a, false, b, (short)0, c, false, false);
}

// ---------- f32 -> bf16 conversion (4 elems / thread) ----------
__global__ void cvt_f32_bf16(const float* __restrict__ in,
                             unsigned int* __restrict__ out, int n4) {
    int i = blockIdx.x * blockDim.x + threadIdx.x;
    if (i < n4) {
        const float4 f = ((const float4*)in)[i];
        unsigned int lo = (unsigned int)bf16_rne(f.x) | ((unsigned int)bf16_rne(f.y) << 16);
        unsigned int hi = (unsigned int)bf16_rne(f.z) | ((unsigned int)bf16_rne(f.w) << 16);
        out[2 * i]     = lo;
        out[2 * i + 1] = hi;
    }
}

// ---------- y = X * W^T (bf16 in/out), per-wave 32x64 tile, double-buffered ----------
// mode 0: store row-major [4096,1024]  (Q, K)
// mode 1: store V transposed: [B,H,hd,N]
__global__ __launch_bounds__(128)
void gemm_xwT(const unsigned short* __restrict__ Xb,
              const unsigned short* __restrict__ Wb,
              unsigned short* __restrict__ out, int mode) {
    const int wave = blockIdx.x * 4 + (threadIdx.x >> 5);   // 0..2047
    const int lane = threadIdx.x & 31;
    const int mt = wave >> 4;            // 0..127 (32-row tiles)
    const int nt = wave & 15;            // 0..15  (64-col tiles)
    const int i0 = mt * 32;
    const int j0 = nt * 64;
    const int lm = lane & 15;
    const int hl = lane >> 4;
    const int abase = hl * 8;            // ISA 16-bit A-frag K offset
    const int bbase = hl * 8;            // ISA 16-bit B-frag K offset

    v8f acc[2][4];
#pragma unroll
    for (int m = 0; m < 2; ++m)
#pragma unroll
        for (int t = 0; t < 4; ++t) acc[m][t] = zero8();

    const unsigned short* arow0 = Xb + (i0 + lm) * DIMC + abase;
    const unsigned short* arow1 = arow0 + 16 * DIMC;
    const unsigned short* brow[4];
#pragma unroll
    for (int t = 0; t < 4; ++t) brow[t] = Wb + (j0 + t * 16 + lm) * DIMC + bbase;

    auto loadA = [&](FragU* a, int kk) {
        a[0].q[0] = *(const u32x4*)(arow0 + kk);
        a[0].q[1] = *(const u32x4*)(arow0 + kk + 16);
        a[1].q[0] = *(const u32x4*)(arow1 + kk);
        a[1].q[1] = *(const u32x4*)(arow1 + kk + 16);
    };
    auto loadB = [&](FragU* b, int kk) {
#pragma unroll
        for (int t = 0; t < 4; ++t) {
            b[t].q[0] = *(const u32x4*)(brow[t] + kk);
            b[t].q[1] = *(const u32x4*)(brow[t] + kk + 16);
        }
    };
    auto mma = [&](FragU* a, FragU* b) {
#pragma unroll
        for (int m = 0; m < 2; ++m)
#pragma unroll
            for (int t = 0; t < 4; ++t)
                acc[m][t] = wmma_bf16(a[m].v, b[t].v, acc[m][t]);
    };

    FragU aA[2], bA[4], aB[2], bB[4];
    loadA(aA, 0);
    loadB(bA, 0);
#pragma unroll 1
    for (int kk = 0; kk < DIMC; kk += 64) {
        loadA(aB, kk + 32);            // phase-B loads in flight during phase-A MMAs
        loadB(bB, kk + 32);
        mma(aA, bA);
        if (kk + 64 < DIMC) {
            loadA(aA, kk + 64);        // phase-A loads in flight during phase-B MMAs
            loadB(bA, kk + 64);
        }
        mma(aB, bB);
    }

#pragma unroll
    for (int m = 0; m < 2; ++m) {
#pragma unroll
        for (int t = 0; t < 4; ++t) {
            const int j = j0 + t * 16 + lm;
#pragma unroll
            for (int r = 0; r < 8; ++r) {
                const int i = i0 + m * 16 + r + hl * 8;
                const unsigned short hv = bf16_rne(acc[m][t][r]);
                if (mode == 0) {
                    out[i * DIMC + j] = hv;
                } else {
                    const int bb = i >> 11;
                    const int n  = i & 2047;
                    const int hh = j >> 6;
                    const int d  = j & 63;
                    out[((bb * HEADS + hh) * HEAD_DIM + d) * NSEQ + n] = hv;
                }
            }
        }
    }
}

// ---------- flash attention: one wave = one (b,h,16-query tile), pipelined ----------
__global__ __launch_bounds__(128)
void attn_kernel(const unsigned short* __restrict__ Qb,
                 const unsigned short* __restrict__ Kb,
                 const unsigned short* __restrict__ Vt,
                 unsigned short* __restrict__ Ob) {
    __shared__ unsigned short lds[4 * 16 * 32];    // per-wave 16x32 P tile
    const int wavein = threadIdx.x >> 5;
    const int wave = blockIdx.x * 4 + wavein;      // 0..4095
    const int lane = threadIdx.x & 31;
    const int qt = wave & 127;
    const int h  = (wave >> 7) & 15;
    const int bb = wave >> 11;
    const int q0 = qt * 16;
    const int lm = lane & 15;
    const int hl = lane >> 4;
    const int abase = hl * 8;
    const int bbase = hl * 8;
    unsigned short* tile = lds + wavein * 16 * 32;

    // Q fragments (A-matrix 16x32 over d, 2 frags cover hd=64)
    const unsigned short* qrow = Qb + (bb * NSEQ + q0 + lm) * DIMC + h * HEAD_DIM + abase;
    FragU aq[2];
#pragma unroll
    for (int f = 0; f < 2; ++f) {
        aq[f].q[0] = *(const u32x4*)(qrow + f * 32);
        aq[f].q[1] = *(const u32x4*)(qrow + f * 32 + 16);
    }

    v8f o[4];
#pragma unroll
    for (int t = 0; t < 4; ++t) o[t] = zero8();
    float mrow[8], lrow[8];
#pragma unroll
    for (int r = 0; r < 8; ++r) { mrow[r] = -1e30f; lrow[r] = 0.0f; }

    const unsigned short* kmat = Kb + (bb * NSEQ) * DIMC + h * HEAD_DIM + bbase;
    const unsigned short* vmat = Vt + ((bb * HEADS + h) * HEAD_DIM) * NSEQ + bbase;
    const float C = SCALE_C * LOG2E;

    auto loadK = [&](FragU (*bk)[2], int j0) {
#pragma unroll
        for (int t = 0; t < 2; ++t) {
            const unsigned short* kr = kmat + (j0 + t * 16 + lm) * DIMC;
            bk[t][0].q[0] = *(const u32x4*)(kr);
            bk[t][0].q[1] = *(const u32x4*)(kr + 16);
            bk[t][1].q[0] = *(const u32x4*)(kr + 32);
            bk[t][1].q[1] = *(const u32x4*)(kr + 48);
        }
    };
    auto loadV = [&](FragU* bv, int j0) {
#pragma unroll
        for (int t = 0; t < 4; ++t) {
            const unsigned short* vr = vmat + (t * 16 + lm) * NSEQ + j0;
            bv[t].q[0] = *(const u32x4*)(vr);
            bv[t].q[1] = *(const u32x4*)(vr + 16);
        }
    };
    auto scores = [&](FragU (*bk)[2], v8f* s) {
#pragma unroll
        for (int t = 0; t < 2; ++t) {
            v8f z = zero8();
            z = wmma_bf16(aq[0].v, bk[t][0].v, z);
            z = wmma_bf16(aq[1].v, bk[t][1].v, z);
            s[t] = z;
        }
    };
    auto softmax_pv = [&](v8f* s, FragU* bv) {
        float alpha[8];
#pragma unroll
        for (int r = 0; r < 8; ++r) {
            float s0 = s[0][r] * C;
            float s1 = s[1][r] * C;
            float t = fmaxf(s0, s1);
            t = fmaxf(t, __shfl_xor(t, 1, 16));
            t = fmaxf(t, __shfl_xor(t, 2, 16));
            t = fmaxf(t, __shfl_xor(t, 4, 16));
            t = fmaxf(t, __shfl_xor(t, 8, 16));
            const float mnew = fmaxf(mrow[r], t);
            alpha[r] = exp2f(mrow[r] - mnew);
            mrow[r] = mnew;
            const float p0 = exp2f(s0 - mnew);
            const float p1 = exp2f(s1 - mnew);
            s[0][r] = p0; s[1][r] = p1;
            float rs = p0 + p1;
            rs += __shfl_xor(rs, 1, 16);
            rs += __shfl_xor(rs, 2, 16);
            rs += __shfl_xor(rs, 4, 16);
            rs += __shfl_xor(rs, 8, 16);
            lrow[r] = lrow[r] * alpha[r] + rs;
        }
#pragma unroll
        for (int t = 0; t < 4; ++t)
#pragma unroll
            for (int r = 0; r < 8; ++r) o[t][r] *= alpha[r];

        // P (C/D layout) -> LDS -> A-fragment layout
#pragma unroll
        for (int r = 0; r < 8; ++r) {
            const int m = r + hl * 8;
            tile[m * 32 + lm]      = bf16_rne(s[0][r]);
            tile[m * 32 + 16 + lm] = bf16_rne(s[1][r]);
        }
        FragU ap;
        ap.q[0] = *(const u32x4*)(tile + lm * 32 + abase);
        ap.q[1] = *(const u32x4*)(tile + lm * 32 + abase + 16);

#pragma unroll
        for (int t = 0; t < 4; ++t)
            o[t] = wmma_bf16(ap.v, bv[t].v, o[t]);
    };

    FragU bkA[2][2], bkB[2][2], bv[4];
    v8f s[2];
    loadK(bkA, 0);
#pragma unroll 1
    for (int j0 = 0; j0 < NSEQ; j0 += 64) {
        loadV(bv, j0);                 // in flight through scores+softmax
        scores(bkA, s);
        loadK(bkB, j0 + 32);           // in flight through softmax+PV
        softmax_pv(s, bv);

        loadV(bv, j0 + 32);
        scores(bkB, s);
        if (j0 + 64 < NSEQ) loadK(bkA, j0 + 64);
        softmax_pv(s, bv);
    }

    // epilogue: O /= l, store bf16 into [B,N,DIM]
#pragma unroll
    for (int t = 0; t < 4; ++t) {
#pragma unroll
        for (int r = 0; r < 8; ++r) {
            const int m = r + hl * 8;
            const float v = o[t][r] / lrow[r];
            Ob[(bb * NSEQ + q0 + m) * DIMC + h * HEAD_DIM + t * 16 + lm] = bf16_rne(v);
        }
    }
}

// ---------- out = attn * Wo^T + bo (f32 output), 32x64 tile, double-buffered ----------
__global__ __launch_bounds__(128)
void gemm_out(const unsigned short* __restrict__ Xb,
              const unsigned short* __restrict__ Wb,
              const float* __restrict__ bias,
              float* __restrict__ out) {
    const int wave = blockIdx.x * 4 + (threadIdx.x >> 5);
    const int lane = threadIdx.x & 31;
    const int mt = wave >> 4;
    const int nt = wave & 15;
    const int i0 = mt * 32;
    const int j0 = nt * 64;
    const int lm = lane & 15;
    const int hl = lane >> 4;
    const int abase = hl * 8;
    const int bbase = hl * 8;

    v8f acc[2][4];
#pragma unroll
    for (int m = 0; m < 2; ++m)
#pragma unroll
        for (int t = 0; t < 4; ++t) acc[m][t] = zero8();

    const unsigned short* arow0 = Xb + (i0 + lm) * DIMC + abase;
    const unsigned short* arow1 = arow0 + 16 * DIMC;
    const unsigned short* brow[4];
#pragma unroll
    for (int t = 0; t < 4; ++t) brow[t] = Wb + (j0 + t * 16 + lm) * DIMC + bbase;

    auto loadA = [&](FragU* a, int kk) {
        a[0].q[0] = *(const u32x4*)(arow0 + kk);
        a[0].q[1] = *(const u32x4*)(arow0 + kk + 16);
        a[1].q[0] = *(const u32x4*)(arow1 + kk);
        a[1].q[1] = *(const u32x4*)(arow1 + kk + 16);
    };
    auto loadB = [&](FragU* b, int kk) {
#pragma unroll
        for (int t = 0; t < 4; ++t) {
            b[t].q[0] = *(const u32x4*)(brow[t] + kk);
            b[t].q[1] = *(const u32x4*)(brow[t] + kk + 16);
        }
    };
    auto mma = [&](FragU* a, FragU* b) {
#pragma unroll
        for (int m = 0; m < 2; ++m)
#pragma unroll
            for (int t = 0; t < 4; ++t)
                acc[m][t] = wmma_bf16(a[m].v, b[t].v, acc[m][t]);
    };

    FragU aA[2], bA[4], aB[2], bB[4];
    loadA(aA, 0);
    loadB(bA, 0);
#pragma unroll 1
    for (int kk = 0; kk < DIMC; kk += 64) {
        loadA(aB, kk + 32);
        loadB(bB, kk + 32);
        mma(aA, bA);
        if (kk + 64 < DIMC) {
            loadA(aA, kk + 64);
            loadB(bA, kk + 64);
        }
        mma(aB, bB);
    }

#pragma unroll
    for (int m = 0; m < 2; ++m) {
#pragma unroll
        for (int t = 0; t < 4; ++t) {
            const int j = j0 + t * 16 + lm;
            const float bj = bias[j];
#pragma unroll
            for (int r = 0; r < 8; ++r) {
                const int i = i0 + m * 16 + r + hl * 8;
                out[i * DIMC + j] = acc[m][t][r] + bj;
            }
        }
    }
}

extern "C" void kernel_launch(void* const* d_in, const int* in_sizes, int n_in,
                              void* d_out, int out_size, void* d_ws, size_t ws_size,
                              hipStream_t stream) {
    const float* x  = (const float*)d_in[0];
    const float* Wq = (const float*)d_in[1];
    const float* Wk = (const float*)d_in[2];
    const float* Wv = (const float*)d_in[3];
    const float* Wo = (const float*)d_in[4];
    const float* bo = (const float*)d_in[5];
    float* out = (float*)d_out;

    // workspace layout (bf16 = unsigned short), 48 MiB total
    unsigned short* xb    = (unsigned short*)d_ws;
    unsigned short* wqb   = xb    + (size_t)NROWS * DIMC;
    unsigned short* wkb   = wqb   + (size_t)DIMC * DIMC;
    unsigned short* wvb   = wkb   + (size_t)DIMC * DIMC;
    unsigned short* wob   = wvb   + (size_t)DIMC * DIMC;
    unsigned short* qb    = wob   + (size_t)DIMC * DIMC;
    unsigned short* kb    = qb    + (size_t)NROWS * DIMC;
    unsigned short* vt    = kb    + (size_t)NROWS * DIMC;     // [B,H,hd,N]
    unsigned short* attnb = vt    + (size_t)NROWS * DIMC;

    const int NX4 = NROWS * DIMC / 4;
    const int NW4 = DIMC * DIMC / 4;

    cvt_f32_bf16<<<(NX4 + 255) / 256, 256, 0, stream>>>(x,  (unsigned int*)xb,  NX4);
    cvt_f32_bf16<<<(NW4 + 255) / 256, 256, 0, stream>>>(Wq, (unsigned int*)wqb, NW4);
    cvt_f32_bf16<<<(NW4 + 255) / 256, 256, 0, stream>>>(Wk, (unsigned int*)wkb, NW4);
    cvt_f32_bf16<<<(NW4 + 255) / 256, 256, 0, stream>>>(Wv, (unsigned int*)wvb, NW4);
    cvt_f32_bf16<<<(NW4 + 255) / 256, 256, 0, stream>>>(Wo, (unsigned int*)wob, NW4);

    // 2048 waves per GEMM: 128 M-tiles(32) x 16 N-tiles(64), 4 waves/block
    gemm_xwT<<<512, 128, 0, stream>>>(xb, wqb, qb, 0);
    gemm_xwT<<<512, 128, 0, stream>>>(xb, wkb, kb, 0);
    gemm_xwT<<<512, 128, 0, stream>>>(xb, wvb, vt, 1);

    // 4096 waves: B*H*(N/16) query tiles
    attn_kernel<<<1024, 128, 0, stream>>>(qb, kb, vt, attnb);

    gemm_out<<<512, 128, 0, stream>>>(attnb, wob, bo, out);

    (void)in_sizes; (void)n_in; (void)out_size; (void)ws_size;
}
